// Mixture_11476152615472
// MI455X (gfx1250) — matrix operand order repaired and checked by reading, not compiled, not run on hardware
//
#include <hip/hip_runtime.h>
#include <hip/hip_bf16.h>
#include <math.h>

typedef _Float16 v16h __attribute__((ext_vector_type(16)));
typedef _Float16 v8h  __attribute__((ext_vector_type(8)));
typedef float    v8f  __attribute__((ext_vector_type(8)));
typedef float    v4f  __attribute__((ext_vector_type(4)));
typedef unsigned int u32x4 __attribute__((ext_vector_type(4)));
typedef int          i32x8 __attribute__((ext_vector_type(8)));
typedef int          i32x4 __attribute__((ext_vector_type(4)));

#define B_   32
#define C_   256
#define HW_  4096
#define F_   256
#define PT_STRIDE 264   // padded row stride (halves) for LDS P^T (512B + 16B pad)
#define ET_STRIDE 264   // padded row stride (halves) for LDS E^T

// LDS layout (bytes)
#define OFF_SX    0                         // float [C_][64]           = 65536
#define OFF_SMAX  65536                     // float [4][64]            = 1024
#define OFF_SM    66560                     // float [64]               = 256
#define OFF_SET   66816                     // half  [64][ET_STRIDE]    = 33792
#define OFF_SPT   100608                    // half  [F_][PT_STRIDE]    = 135168
#define SMEM_BYTES 235776

#define HAVE_TDM __has_builtin(__builtin_amdgcn_tensor_load_to_lds)

// ---------------------------------------------------------------------------
// Kernel 1: p[c,f] = b[c,f]^2 / sum_c b[c,f]^2, stored transposed as f16:
// PT[f*256 + c]. One block per filter f, one thread per channel c.
// ---------------------------------------------------------------------------
__global__ __launch_bounds__(256) void mix_prep_kernel(
    const float* __restrict__ biases, _Float16* __restrict__ PT) {
  int f = blockIdx.x;
  int c = threadIdx.x;
  float v = biases[c * F_ + f];
  v *= v;
  __shared__ float red[256];
  red[c] = v;
  __syncthreads();
  for (int s = 128; s > 0; s >>= 1) {
    if (c < s) red[c] += red[c + s];
    __syncthreads();
  }
  float sum = red[0];
  PT[f * C_ + c] = (_Float16)(v / sum);
}

#if HAVE_TDM
// Issue a 2D TDM tile load: global (row-major, stride0 elems/row) -> LDS.
// w0flags carries data_size + pad controls (group1 word0, workgroup_mask=0).
static __device__ __forceinline__ void tdm_load_2d(
    unsigned lds_off, const void* gaddr, unsigned w0flags,
    unsigned td0, unsigned td1, unsigned tile0, unsigned tile1,
    unsigned long long stride0) {
  unsigned long long ga = (unsigned long long)(uintptr_t)gaddr;
  u32x4 g0;
  g0[0] = 1u;                                   // count=1 (valid), user mode
  g0[1] = lds_off;                              // lds_addr (bytes)
  g0[2] = (unsigned)ga;                         // global_addr[31:0]
  g0[3] = (unsigned)((ga >> 32) & 0x1FFFFFFu)   // global_addr[56:32]
          | 0x80000000u;                        // type=2 ("image")
  i32x8 g1;
  g1[0] = (int)w0flags;                                        // mask=0|dsize|pad
  g1[1] = (int)((td0 & 0xFFFFu) << 16);                        // tdim0 lo
  g1[2] = (int)(((td0 >> 16) & 0xFFFFu) | ((td1 & 0xFFFFu) << 16));
  g1[3] = (int)(((td1 >> 16) & 0xFFFFu) | ((tile0 & 0xFFFFu) << 16));
  g1[4] = (int)(tile1 & 0xFFFFu);                              // tile_dim1, tile_dim2=0
  g1[5] = (int)(unsigned)(stride0 & 0xFFFFFFFFull);            // stride0 lo
  g1[6] = (int)(unsigned)((stride0 >> 32) & 0xFFFFull);        // stride0 hi, stride1=0
  g1[7] = 0;
  i32x4 gz = {0, 0, 0, 0};
#if defined(__clang_major__) && (__clang_major__ >= 23)
  i32x8 gz8 = {0, 0, 0, 0, 0, 0, 0, 0};
  __builtin_amdgcn_tensor_load_to_lds(g0, g1, gz, gz, gz8, 0);
#else
  __builtin_amdgcn_tensor_load_to_lds(g0, g1, gz, gz, 0);
#endif
}
#endif

static __device__ __forceinline__ void wait_tensorcnt0() {
#if __has_builtin(__builtin_amdgcn_s_wait_tensorcnt)
  __builtin_amdgcn_s_wait_tensorcnt(0);
#else
  asm volatile("s_wait_tensorcnt 0x0" ::: "memory");
#endif
}

// ---------------------------------------------------------------------------
// Kernel 2: fused max / exp / GEMM(WMMA f16->f32) / log epilogue.
// Grid: B_ * (HW_/64) blocks of 256 threads (8 wave32).
// Block handles one batch b and 64 consecutive spatial positions; computes
// out[b, 0..255, pos0..pos0+63]. Input strip and P^T are DMA'd by the TDM.
// ---------------------------------------------------------------------------
__global__ __launch_bounds__(256, 1) void mix_main_kernel(
    const float* __restrict__ in, const _Float16* __restrict__ PTg,
    float* __restrict__ out) {
  extern __shared__ char smem[];
  float*    sX   = reinterpret_cast<float*>(smem + OFF_SX);     // [c][pos]
  float*    sMax = reinterpret_cast<float*>(smem + OFF_SMAX);   // [cg][pos]
  float*    sM   = reinterpret_cast<float*>(smem + OFF_SM);     // [pos]
  _Float16* sET  = reinterpret_cast<_Float16*>(smem + OFF_SET); // [pos][c]
  _Float16* sPT  = reinterpret_cast<_Float16*>(smem + OFF_SPT); // [f][c] padded

  const int t    = threadIdx.x;
  const int lane = t & 31;
  const int wave = t >> 5;
  const int b    = blockIdx.x >> 6;        // HW_/64 = 64 tiles per batch
  const int tile = blockIdx.x & 63;
  const int pos0 = tile * 64;

  const int pos    = t & 63;               // phase 1b/2 mapping
  const int cchunk = (t >> 6) * 64;

  // ---- Phase 1: TDM DMA of the input strip and padded P^T into LDS --------
#if HAVE_TDM
  if (wave == 0) {
    // x[b, 0..255, pos0..pos0+63]: 2D tile 64 x 256 of f32, row stride HW_.
    tdm_load_2d(OFF_SX, in + (size_t)b * C_ * HW_ + pos0,
                /*w0flags=*/(2u << 16),            // data_size=4B
                /*td0=*/64, /*td1=*/C_, /*tile0=*/64, /*tile1=*/C_,
                /*stride0=*/(unsigned long long)HW_);
    wait_tensorcnt0();
  } else if (wave == 1) {
    // P^T: 256 rows x 512B, DMA-padded +16B per row -> 528B LDS row stride.
    tdm_load_2d(OFF_SPT, PTg,
                /*w0flags=*/(3u << 16)             // data_size=8B
                           | (1u << 20)            // pad_enable
                           | (6u << 22)            // pad_interval: 128 dwords
                           | (3u << 25),           // pad_amount: 4 dwords (16B)
                /*td0=*/64, /*td1=*/F_, /*tile0=*/64, /*tile1=*/F_,
                /*stride0=*/64ull);
    wait_tensorcnt0();
  }
#else
  // Fallback: manual staging.
  {
    const float* gbase = in + (size_t)b * C_ * HW_ + pos0 + pos;
    for (int j = 0; j < 64; ++j) {
      int c = cchunk + j;
      sX[c * 64 + pos] = gbase[(size_t)c * HW_];
    }
    for (int i = t; i < F_ * 32; i += 256) {
      int row = i >> 5;
      int col = (i & 31) * 8;
      *reinterpret_cast<v8h*>(sPT + row * PT_STRIDE + col) =
          *reinterpret_cast<const v8h*>(PTg + row * C_ + col);
    }
  }
#endif
  __syncthreads();

  // ---- Phase 1b: partial per-position max over this thread's 64 channels --
  {
    float mr = -INFINITY;
#pragma unroll 8
    for (int j = 0; j < 64; ++j)
      mr = fmaxf(mr, sX[(cchunk + j) * 64 + pos]);
    sMax[(t >> 6) * 64 + pos] = mr;
  }
  __syncthreads();

  // ---- Phase 2: m = max_c, e = exp(x-m) -> sET f16 [pos][c] ---------------
  {
    float m = fmaxf(fmaxf(sMax[pos], sMax[64 + pos]),
                    fmaxf(sMax[128 + pos], sMax[192 + pos]));
    if (t < 64) sM[t] = m;
    for (int j0 = 0; j0 < 64; j0 += 8) {
      v8h buf;
#pragma unroll
      for (int u = 0; u < 8; ++u) {
        float x = sX[(cchunk + j0 + u) * 64 + pos];
        buf[u] = (_Float16)__expf(x - m);
      }
      *reinterpret_cast<v8h*>(sET + pos * ET_STRIDE + cchunk + j0) = buf;
    }
  }
  __syncthreads();

  // ---- Phase 3: WMMA GEMM  D[pos,f] = E[pos,c] x P[c,f], out = m + log ----
  // A (16x32 f16): lanes<16: halves0-7 -> K+0..7, halves8-15 -> K+16..23
  //                lanes>=16: halves0-7 -> K+8..15, halves8-15 -> K+24..31
  // B (32x16 f16): same K order as A
  const int n16 = lane & 15;
  const int hiA = (lane >= 16) ? 8 : 0;
  const int hiB = (lane >= 16) ? 8 : 0;
  const int hiM = (lane >= 16) ? 8 : 0;

  union V16 { v16h v; v8h h[2]; };

#pragma unroll
  for (int ft = 0; ft < 2; ++ft) {
    const int fBase = wave * 32 + ft * 16;
    // Hoist the 8 B fragments for this f-tile (all K) into registers.
    V16 bf[8];
    {
      const _Float16* prow = sPT + (fBase + n16) * PT_STRIDE + hiB;
#pragma unroll
      for (int k = 0; k < 8; ++k) {
        bf[k].h[0] = *reinterpret_cast<const v8h*>(prow + k * 32);
        bf[k].h[1] = *reinterpret_cast<const v8h*>(prow + k * 32 + 16);
      }
    }
#pragma unroll
    for (int pt = 0; pt < 4; ++pt) {
      v8f acc = {};
      const _Float16* erow = sET + (pt * 16 + n16) * ET_STRIDE + hiA;
#pragma unroll
      for (int k = 0; k < 8; ++k) {
        V16 af;
        af.h[0] = *reinterpret_cast<const v8h*>(erow + k * 32);
        af.h[1] = *reinterpret_cast<const v8h*>(erow + k * 32 + 16);
        acc = __builtin_amdgcn_wmma_f32_16x16x32_f16(
            false, af.v, false, bf[k].v, (short)0, acc, false, false);
      }
      // Epilogue: out[b, f, pos] = m[pos] + log(acc)
      float* obase = out + ((size_t)(b * F_ + fBase + n16)) * HW_ +
                     pos0 + pt * 16 + hiM;
      v4f r0, r1;
#pragma unroll
      for (int u = 0; u < 4; ++u)
        r0[u] = sM[pt * 16 + hiM + u] + __logf(acc[u]);
#pragma unroll
      for (int u = 0; u < 4; ++u)
        r1[u] = sM[pt * 16 + hiM + 4 + u] + __logf(acc[4 + u]);
      *reinterpret_cast<v4f*>(obase)     = r0;
      *reinterpret_cast<v4f*>(obase + 4) = r1;
    }
  }
}

// ---------------------------------------------------------------------------
extern "C" void kernel_launch(void* const* d_in, const int* in_sizes, int n_in,
                              void* d_out, int out_size, void* d_ws,
                              size_t ws_size, hipStream_t stream) {
  const float* inputs = reinterpret_cast<const float*>(d_in[0]); // (32,256,64,64)
  const float* biases = reinterpret_cast<const float*>(d_in[1]); // (1,1,256,256)
  float* out = reinterpret_cast<float*>(d_out);
  _Float16* PT = reinterpret_cast<_Float16*>(d_ws);              // 128 KB

  (void)in_sizes; (void)n_in; (void)out_size; (void)ws_size;

  // Allow >64KB dynamic LDS for the main kernel (no-op if already set).
  (void)hipFuncSetAttribute(
      reinterpret_cast<const void*>(&mix_main_kernel),
      hipFuncAttributeMaxDynamicSharedMemorySize, SMEM_BYTES);

  mix_prep_kernel<<<dim3(F_), dim3(256), 0, stream>>>(biases, PT);

  mix_main_kernel<<<dim3(B_ * (HW_ / 64)), dim3(256), SMEM_BYTES, stream>>>(
      inputs, PT, out);
}
